// InfiniAttention_42623255445540
// MI455X (gfx1250) — matrix operand order repaired and checked by reading, not compiled, not run on hardware
//
#include <hip/hip_runtime.h>
#include <hip/hip_bf16.h>
#include <math.h>

// Problem constants (from reference setup_inputs)
#define BB 2
#define SS 2048
#define DD 2048
#define HH 16
#define DK 128
#define BH (BB*HH)   // 32
#define BS (BB*SS)   // 4096
#define D3 (3*DD)    // 6144

typedef __attribute__((ext_vector_type(16))) __bf16 v16bf;
typedef __attribute__((ext_vector_type(8)))  __bf16 v8bf;
typedef __attribute__((ext_vector_type(8)))  float  v8f;

static __device__ __forceinline__ v8f wmma_bf16(v16bf a, v16bf b, v8f c){
  // D(16x16,f32) = A(16x32,bf16) * B(32x16,bf16) + C
  return __builtin_amdgcn_wmma_f32_16x16x32_bf16(false, a, false, b, (short)0, c, false, false);
}

static __device__ __forceinline__ v8f zero_v8f(){
  v8f z;
#pragma unroll
  for (int i = 0; i < 8; ++i) z[i] = 0.f;
  return z;
}

// A fragment (16x32 bf16), A row-major (M,K). ISA layout:
// lanes 0-15: halves 0-7 -> K = k0+0..7,  halves 8-15 -> K = k0+16..23
// lanes 16-31: halves 0-7 -> K = k0+8..15, halves 8-15 -> K = k0+24..31
static __device__ __forceinline__ v16bf load_a_frag(const __bf16* __restrict__ A, size_t lda,
                                                    int m0, int k0, int lane){
  int m  = m0 + (lane & 15);
  int hi = lane >> 4;
  const __bf16* p = A + (size_t)m * lda + (size_t)k0 + hi * 8;
  v8bf x = *(const v8bf*)p;          // 16B load
  v8bf y = *(const v8bf*)(p + 16);   // 16B load
  v16bf r;
#pragma unroll
  for (int i = 0; i < 8; ++i){ r[i] = x[i]; r[i + 8] = y[i]; }
  return r;
}

// B fragment (32x16 bf16), supplied as Bt row-major (N,K).
// lanes 0-15: K = k0+0..7, k0+16..23 of column n; lanes 16-31: K = k0+8..15, k0+24..31.
static __device__ __forceinline__ v16bf load_bt_frag(const __bf16* __restrict__ Bt, size_t ldb,
                                                     int n0, int k0, int lane){
  int n  = n0 + (lane & 15);
  int hi = lane >> 4;
  const __bf16* p = Bt + (size_t)n * ldb + (size_t)k0 + hi * 8;
  v8bf x = *(const v8bf*)p;          // 16B load
  v8bf y = *(const v8bf*)(p + 16);   // 16B load
  v16bf r;
#pragma unroll
  for (int i = 0; i < 8; ++i){ r[i] = x[i]; r[i + 8] = y[i]; }
  return r;
}

static __device__ __forceinline__ float elu1f(float x){
  return x > 0.f ? x + 1.f : __expf(x);
}

// ---------------- casts ----------------
__global__ void cast_f32_bf16(const float* __restrict__ src, __bf16* __restrict__ dst, int n){
  int i = blockIdx.x * blockDim.x + threadIdx.x;
  if (i < n) dst[i] = (__bf16)src[i];
}

// Mt[bh][v][k] = M[bh][k][v]  (bf16)
__global__ void cast_M_T(const float* __restrict__ M, __bf16* __restrict__ Mt){
  int i = blockIdx.x * blockDim.x + threadIdx.x;       // BH*DK*DK
  int bh = i >> 14; int rem = i & 16383; int k = rem >> 7; int v = rem & 127;
  Mt[((size_t)bh * DK + v) * DK + k] = (__bf16)M[i];
}

// ---------------- GEMM1: QKV = x @ Wqkv^T, split into head-major Q/K (bf16) and V (f32) ----------------
__global__ __launch_bounds__(256) void gemm_qkv(const __bf16* __restrict__ xb, const __bf16* __restrict__ wqb,
                                                __bf16* __restrict__ Qb, __bf16* __restrict__ Kb,
                                                float* __restrict__ Vf){
  int lane = threadIdx.x & 31, wave = threadIdx.x >> 5;
  int m0 = blockIdx.x * 128 + wave * 16;
  int n0 = blockIdx.y * 64;
  v8f c[4]; 
#pragma unroll
  for (int t = 0; t < 4; ++t) c[t] = zero_v8f();
  for (int k0 = 0; k0 < DD; k0 += 32){
    v16bf a = load_a_frag(xb, DD, m0, k0, lane);
#pragma unroll
    for (int t = 0; t < 4; ++t)
      c[t] = wmma_bf16(a, load_bt_frag(wqb, DD, n0 + 16 * t, k0, lane), c[t]);
  }
  int hi = lane >> 4, ln = lane & 15;
#pragma unroll
  for (int t = 0; t < 4; ++t)
#pragma unroll
    for (int r = 0; r < 8; ++r){
      int m = m0 + r + 8 * hi;
      int e = n0 + 16 * t + ln;
      int bi = m / SS, s = m % SS;
      int sec = e / DD, col = e % DD, h = col >> 7, d = col & 127;
      size_t o = (((size_t)(bi * HH + h)) * SS + s) * DK + d;
      float v = c[t][r];
      if (sec == 0)      Qb[o] = (__bf16)v;
      else if (sec == 1) Kb[o] = (__bf16)v;
      else               Vf[o] = v;
    }
}

// ---------------- phi / denominators / transposed copies ----------------
__global__ __launch_bounds__(256) void phi_kernel(const __bf16* __restrict__ Qb, const __bf16* __restrict__ Kb,
                                                  const float* __restrict__ Vf, const float* __restrict__ z,
                                                  __bf16* __restrict__ phiQ, __bf16* __restrict__ phiK,
                                                  __bf16* __restrict__ phiKT, __bf16* __restrict__ VT,
                                                  float* __restrict__ denq, float* __restrict__ denk){
  int lane = threadIdx.x & 31, wave = threadIdx.x >> 5;
  int row = blockIdx.x * 8 + wave;      // bh*S + s
  int bh = row >> 11, s = row & 2047;
  size_t base = (size_t)row * DK;
  float dq = 0.f, dkk = 0.f;
#pragma unroll
  for (int j = 0; j < 4; ++j){
    int d = lane + 32 * j;
    float zv = z[bh * DK + d];
    float pq = elu1f((float)Qb[base + d]);
    float pk = elu1f((float)Kb[base + d]);
    phiQ[base + d] = (__bf16)pq;
    phiK[base + d] = (__bf16)pk;
    size_t tidx = ((size_t)bh * DK + d) * SS + s;
    phiKT[tidx] = (__bf16)pk;
    VT[tidx]    = (__bf16)Vf[base + d];
    dq  += pq * zv;
    dkk += pk * zv;
  }
#pragma unroll
  for (int m = 16; m >= 1; m >>= 1){
    dq  += __shfl_xor(dq,  m, 32);
    dkk += __shfl_xor(dkk, m, 32);
  }
  if (lane == 0){
    denq[row] = fmaxf(dq,  1e-6f);
    denk[row] = fmaxf(dkk, 1e-6f);
  }
}

// ---------------- GEMM2/3: phi @ M ; mode 0 -> A_mem (B,S,D layout); mode 1 -> U^T bf16 ----------------
__global__ __launch_bounds__(256) void gemm_phiM(const __bf16* __restrict__ phi, const __bf16* __restrict__ Mt,
                                                 const float* __restrict__ den, const float* __restrict__ Vf,
                                                 float* __restrict__ Amem, __bf16* __restrict__ UT, int mode){
  int lane = threadIdx.x & 31, wave = threadIdx.x >> 5;
  int bh = blockIdx.z; int bi = bh >> 4; int h = bh & 15;
  int m0 = blockIdx.x * 128 + wave * 16;   // s tile
  int n0 = blockIdx.y * 64;                // output-dim tile
  const __bf16* A  = phi + (size_t)bh * SS * DK;
  const __bf16* Bt = Mt  + (size_t)bh * DK * DK;
  v8f c[4];
#pragma unroll
  for (int t = 0; t < 4; ++t) c[t] = zero_v8f();
  for (int k0 = 0; k0 < DK; k0 += 32){
    v16bf a = load_a_frag(A, DK, m0, k0, lane);
#pragma unroll
    for (int t = 0; t < 4; ++t)
      c[t] = wmma_bf16(a, load_bt_frag(Bt, DK, n0 + 16 * t, k0, lane), c[t]);
  }
  int hi = lane >> 4, ln = lane & 15;
#pragma unroll
  for (int t = 0; t < 4; ++t)
#pragma unroll
    for (int r = 0; r < 8; ++r){
      int s = m0 + r + 8 * hi;
      int n = n0 + 16 * t + ln;
      float val = c[t][r] / den[bh * SS + s];
      if (mode == 0){
        Amem[((size_t)(bi * SS + s)) * DD + h * DK + n] = val;
      } else {
        float u = Vf[((size_t)bh * SS + s) * DK + n] - val;
        UT[((size_t)bh * DK + n) * SS + s] = (__bf16)u;
      }
    }
}

// ---------------- GEMM4: M_new = M + phiK^T @ U  (computed as U^T @ phiK, stored transposed) ----------------
__global__ __launch_bounds__(256) void gemm_mdelta(const __bf16* __restrict__ UT, const __bf16* __restrict__ phiKT,
                                                   const float* __restrict__ Min, float* __restrict__ Mout){
  int lane = threadIdx.x & 31, wave = threadIdx.x >> 5;
  int bh = blockIdx.z;
  int m0 = wave * 16;             // v index
  int n0 = blockIdx.y * 64;       // k index
  const __bf16* A  = UT    + (size_t)bh * DK * SS;
  const __bf16* Bt = phiKT + (size_t)bh * DK * SS;
  v8f c[4];
#pragma unroll
  for (int t = 0; t < 4; ++t) c[t] = zero_v8f();
  for (int k0 = 0; k0 < SS; k0 += 32){
    v16bf a = load_a_frag(A, SS, m0, k0, lane);
#pragma unroll
    for (int t = 0; t < 4; ++t)
      c[t] = wmma_bf16(a, load_bt_frag(Bt, SS, n0 + 16 * t, k0, lane), c[t]);
  }
  int hi = lane >> 4, ln = lane & 15;
#pragma unroll
  for (int t = 0; t < 4; ++t)
#pragma unroll
    for (int r = 0; r < 8; ++r){
      int v = m0 + r + 8 * hi;
      int k = n0 + 16 * t + ln;
      size_t o = ((size_t)bh * DK + k) * DK + v;
      Mout[o] = Min[o] + c[t][r];
    }
}

// ---------------- z_new = z + sum_s elu1(K) ----------------
__global__ void znew_kernel(const __bf16* __restrict__ Kb, const float* __restrict__ z,
                            float* __restrict__ zout){
  int bh = blockIdx.x; int d = threadIdx.x;   // 128 threads
  const __bf16* p = Kb + (size_t)bh * SS * DK + d;
  float acc = 0.f;
  for (int s = 0; s < SS; ++s)
    acc += elu1f((float)p[(size_t)s * DK]);
  zout[bh * DK + d] = z[bh * DK + d] + acc;
}

// ---------------- Flash attention (causal) + gated combine with A_mem ----------------
__global__ __launch_bounds__(256) void flash_kernel(const __bf16* __restrict__ Qb, const __bf16* __restrict__ Kb,
                                                    const __bf16* __restrict__ VT, const float* __restrict__ Amem,
                                                    const float* __restrict__ beta, __bf16* __restrict__ Acomb){
  __shared__ __bf16 pLds[8][16][32];   // per-wave P tile (16 q-rows x 32 k-cols), 8 KB total
  int lane = threadIdx.x & 31, wave = threadIdx.x >> 5;
  int hi = lane >> 4, ln = lane & 15;
  int bh = blockIdx.y, bi = bh >> 4, h = bh & 15;
  int qBase = blockIdx.x * 128 + wave * 16;
  const __bf16* Qbh = Qb + (size_t)bh * SS * DK;
  const __bf16* Kbh = Kb + (size_t)bh * SS * DK;
  const __bf16* Vbh = VT + (size_t)bh * DK * SS;

  v16bf aq[4];
#pragma unroll
  for (int j = 0; j < 4; ++j) aq[j] = load_a_frag(Qbh, DK, qBase, j * 32, lane);

  v8f o[8];
#pragma unroll
  for (int nt = 0; nt < 8; ++nt) o[nt] = zero_v8f();
  float mrow[8], lrow[8];
#pragma unroll
  for (int r = 0; r < 8; ++r){ mrow[r] = -3.0e38f; lrow[r] = 0.f; }

  const float scale = 0.088388347648318447f;  // 1/sqrt(128)

  for (int s0 = 0; s0 <= qBase + 15; s0 += 32){
    v8f c0 = zero_v8f(), c1 = zero_v8f();
#pragma unroll
    for (int j = 0; j < 4; ++j){
      c0 = wmma_bf16(aq[j], load_bt_frag(Kbh, DK, s0,      j * 32, lane), c0);
      c1 = wmma_bf16(aq[j], load_bt_frag(Kbh, DK, s0 + 16, j * 32, lane), c1);
    }
    int col0 = s0 + ln, col1 = s0 + 16 + ln;
#pragma unroll
    for (int r = 0; r < 8; ++r){
      int q = qBase + r + 8 * hi;
      float x0 = (col0 <= q) ? c0[r] * scale : -3.0e38f;
      float x1 = (col1 <= q) ? c1[r] * scale : -3.0e38f;
      float mx = fmaxf(x0, x1);
#pragma unroll
      for (int msk = 1; msk < 16; msk <<= 1) mx = fmaxf(mx, __shfl_xor(mx, msk, 32));
      float mnew = fmaxf(mrow[r], mx);
      float corr = __expf(mrow[r] - mnew);
      float e0 = __expf(x0 - mnew);
      float e1 = __expf(x1 - mnew);
      float sum = e0 + e1;
#pragma unroll
      for (int msk = 1; msk < 16; msk <<= 1) sum += __shfl_xor(sum, msk, 32);
      lrow[r] = lrow[r] * corr + sum;
      mrow[r] = mnew;
#pragma unroll
      for (int nt = 0; nt < 8; ++nt) o[nt][r] *= corr;
      pLds[wave][r + 8 * hi][ln]      = (__bf16)e0;
      pLds[wave][r + 8 * hi][16 + ln] = (__bf16)e1;
    }
    // Re-shape P tile (C layout) into an A fragment via this wave's private LDS slab.
    v16bf pa;
    {
      const __bf16* p = &pLds[wave][ln][0];
      v8bf x = *(const v8bf*)(p + hi * 8);
      v8bf y = *(const v8bf*)(p + 16 + hi * 8);
#pragma unroll
      for (int i = 0; i < 8; ++i){ pa[i] = x[i]; pa[i + 8] = y[i]; }
    }
#pragma unroll
    for (int nt = 0; nt < 8; ++nt)
      o[nt] = wmma_bf16(pa, load_bt_frag(Vbh, SS, nt * 16, s0, lane), o[nt]);
  }

  float g = 1.f / (1.f + __expf(-beta[h]));
#pragma unroll
  for (int nt = 0; nt < 8; ++nt)
#pragma unroll
    for (int r = 0; r < 8; ++r){
      int q = qBase + r + 8 * hi;
      int v = nt * 16 + ln;
      size_t idx = ((size_t)(bi * SS + q)) * DD + h * DK + v;
      float sd = o[nt][r] / lrow[r];
      Acomb[idx] = (__bf16)(g * Amem[idx] + (1.f - g) * sd);
    }
}

// ---------------- GEMM5: out = Acomb @ Wout^T (f32 out) ----------------
__global__ __launch_bounds__(256) void gemm_out(const __bf16* __restrict__ Ab, const __bf16* __restrict__ Wb,
                                                float* __restrict__ out){
  int lane = threadIdx.x & 31, wave = threadIdx.x >> 5;
  int m0 = blockIdx.x * 128 + wave * 16;
  int n0 = blockIdx.y * 64;
  v8f c[4];
#pragma unroll
  for (int t = 0; t < 4; ++t) c[t] = zero_v8f();
  for (int k0 = 0; k0 < DD; k0 += 32){
    v16bf a = load_a_frag(Ab, DD, m0, k0, lane);
#pragma unroll
    for (int t = 0; t < 4; ++t)
      c[t] = wmma_bf16(a, load_bt_frag(Wb, DD, n0 + 16 * t, k0, lane), c[t]);
  }
  int hi = lane >> 4, ln = lane & 15;
#pragma unroll
  for (int t = 0; t < 4; ++t)
#pragma unroll
    for (int r = 0; r < 8; ++r){
      int m = m0 + r + 8 * hi;
      int n = n0 + 16 * t + ln;
      out[(size_t)m * DD + n] = c[t][r];
    }
}

extern "C" void kernel_launch(void* const* d_in, const int* in_sizes, int n_in,
                              void* d_out, int out_size, void* d_ws, size_t ws_size,
                              hipStream_t stream){
  (void)in_sizes; (void)n_in; (void)out_size; (void)ws_size;
  const float* x    = (const float*)d_in[0];
  const float* M    = (const float*)d_in[1];
  const float* z    = (const float*)d_in[2];
  const float* Wqkv = (const float*)d_in[3];
  const float* Wout = (const float*)d_in[4];
  const float* beta = (const float*)d_in[5];

  float* out  = (float*)d_out;
  float* Mout = out + (size_t)BS * DD;
  float* zout = Mout + (size_t)BH * DK * DK;

  char* w = (char*)d_ws;
  size_t off = 0;
  auto alloc = [&](size_t bytes) -> char* {
    char* p = w + off;
    off = (off + bytes + 255) & ~(size_t)255;
    return p;
  };
  __bf16* xb    = (__bf16*)alloc((size_t)BS * DD * 2);
  __bf16* wqb   = (__bf16*)alloc((size_t)D3 * DD * 2);
  __bf16* wob   = (__bf16*)alloc((size_t)DD * DD * 2);
  __bf16* Qb    = (__bf16*)alloc((size_t)BH * SS * DK * 2);
  __bf16* Kb    = (__bf16*)alloc((size_t)BH * SS * DK * 2);
  float*  Vf    = (float*) alloc((size_t)BH * SS * DK * 4);
  __bf16* Mt    = (__bf16*)alloc((size_t)BH * DK * DK * 2);
  __bf16* phiQ  = (__bf16*)alloc((size_t)BH * SS * DK * 2);
  __bf16* phiK  = (__bf16*)alloc((size_t)BH * SS * DK * 2);
  __bf16* phiKT = (__bf16*)alloc((size_t)BH * SS * DK * 2);
  __bf16* VT    = (__bf16*)alloc((size_t)BH * SS * DK * 2);
  __bf16* UT    = (__bf16*)alloc((size_t)BH * SS * DK * 2);
  float*  denq  = (float*) alloc((size_t)BH * SS * 4);
  float*  denk  = (float*) alloc((size_t)BH * SS * 4);
  float*  Amem  = (float*) alloc((size_t)BS * DD * 4);
  __bf16* Acomb = (__bf16*)alloc((size_t)BS * DD * 2);

  cast_f32_bf16<<<(BS * DD) / 256, 256, 0, stream>>>(x, xb, BS * DD);
  cast_f32_bf16<<<(D3 * DD) / 256, 256, 0, stream>>>(Wqkv, wqb, D3 * DD);
  cast_f32_bf16<<<(DD * DD) / 256, 256, 0, stream>>>(Wout, wob, DD * DD);
  cast_M_T<<<(BH * DK * DK) / 256, 256, 0, stream>>>(M, Mt);

  gemm_qkv<<<dim3(BS / 128, D3 / 64), 256, 0, stream>>>(xb, wqb, Qb, Kb, Vf);
  phi_kernel<<<(BH * SS) / 8, 256, 0, stream>>>(Qb, Kb, Vf, z, phiQ, phiK, phiKT, VT, denq, denk);
  gemm_phiM<<<dim3(SS / 128, DK / 64, BH), 256, 0, stream>>>(phiQ, Mt, denq, Vf, Amem, UT, 0);
  gemm_phiM<<<dim3(SS / 128, DK / 64, BH), 256, 0, stream>>>(phiK, Mt, denk, Vf, Amem, UT, 1);
  gemm_mdelta<<<dim3(1, DK / 64, BH), 256, 0, stream>>>(UT, phiKT, M, Mout);
  znew_kernel<<<BH, DK, 0, stream>>>(Kb, z, zout);
  flash_kernel<<<dim3(SS / 128, BH), 256, 0, stream>>>(Qb, Kb, VT, Amem, beta, Acomb);
  gemm_out<<<dim3(BS / 128, DD / 64), 256, 0, stream>>>(Acomb, wob, out);
}
